// GNNEncoder_43636867727540
// MI455X (gfx1250) — compile-verified, matrix operand order checked by reading, not run on hardware
//
#include <hip/hip_runtime.h>
#include <hip/hip_bf16.h>
#include <cstdint>

typedef __attribute__((ext_vector_type(16))) __bf16 v16bf;
typedef __attribute__((ext_vector_type(8)))  float  v8f;
typedef __attribute__((ext_vector_type(4)))  int    v4i;

#define N_NEWS 100000
#define N_KW   20000
#define N_ST   5000
#define D_NEWS 768
#define D_KW   128
#define D_ST   64
#define HDIM   512
#define ODIM   256
#define E_HK   500000
#define E_HS   200000

#define BM 128
#define BN 128
#define BK 32

#if (__has_builtin(__builtin_amdgcn_global_load_async_to_lds_b128) && \
     __has_builtin(__builtin_amdgcn_s_wait_asynccnt))
#define HAS_ASYNC_LDS 1
#else
#define HAS_ASYNC_LDS 0
#endif

// b128 async-copy builtin takes int4* in AS(1)/AS(3)
#define AS1V4(p) ((__attribute__((address_space(1))) v4i*)(uintptr_t)(p))
#define AS3V4(p) ((__attribute__((address_space(3))) v4i*)(unsigned)(uintptr_t)(p))

// ---------------------------------------------------------------- elementwise
__global__ void gnn_zero_f32(float* __restrict__ p, long n) {
  long i = (long)blockIdx.x * blockDim.x + threadIdx.x;
  if (i < n) p[i] = 0.0f;
}

__global__ void gnn_cvt_bf16(const float* __restrict__ in, __bf16* __restrict__ out, long n) {
  long i = (long)blockIdx.x * blockDim.x + threadIdx.x;
  if (i < n) out[i] = (__bf16)in[i];
}

__global__ void gnn_relu_cvt(const float* __restrict__ in, __bf16* __restrict__ out, long n) {
  long i = (long)blockIdx.x * blockDim.x + threadIdx.x;
  if (i < n) out[i] = (__bf16)fmaxf(in[i], 0.0f);
}

// agg f32 / max(cnt,1) -> bf16 mean  (rows are HDIM=512 wide)
__global__ void gnn_mean_cvt(const float* __restrict__ agg, const float* __restrict__ cnt,
                             __bf16* __restrict__ out, long total) {
  long i = (long)blockIdx.x * blockDim.x + threadIdx.x;
  if (i < total) {
    long row = i >> 9;  // /512
    out[i] = (__bf16)(agg[i] / fmaxf(cnt[row], 1.0f));
  }
}

// ---------------------------------------------------------------- edge scatter
// one wave32 per edge; 512-dim bf16 features gathered, f32 atomics accumulated
__global__ void gnn_scatter(const __bf16* __restrict__ feat, const int* __restrict__ src,
                            const int* __restrict__ dst, float* __restrict__ agg,
                            float* __restrict__ cnt, int E) {
  int edge = blockIdx.x * (blockDim.x >> 5) + (threadIdx.x >> 5);
  int lane = threadIdx.x & 31;
  if (edge >= E) return;
  int s = src[edge];
  int d = dst[edge];
  const __bf16* fp = feat + (size_t)s * HDIM;
  float* ap = agg + (size_t)d * HDIM;
#pragma unroll
  for (int i = 0; i < 2; ++i) {
    int base = lane * 16 + i * 8;  // element offset, 16B-aligned chunks
    union { __bf16 h[8]; uint4 q; } u;
    u.q = *(const uint4*)(fp + base);
#pragma unroll
    for (int j = 0; j < 8; ++j) atomicAdd(ap + base + j, (float)u.h[j]);
  }
  if (lane == 0) atomicAdd(cnt + d, 1.0f);
}

// ---------------------------------------------------------------- WMMA GEMM
// C[M,N] = op( A[M,K](bf16) @ W[N,K]^T(bf16) + bias + beta*Cf ), f32 accum.
// Block tile 128x128, 8 waves (4 M x 2 N), wave tile 32x64 = 2x4 WMMA tiles.
// Double-buffered LDS staging via global_load_async_to_lds_b128 (4 async ops /
// wave / tile, uniform: OOB rows clamp to M-1; garbage only reaches masked rows).
__launch_bounds__(256)
__global__ void gnn_gemm(const __bf16* __restrict__ A, const __bf16* __restrict__ W,
                         const float* __restrict__ bias,
                         float* __restrict__ Cf, __bf16* __restrict__ Cb,
                         int M, int N, int K, int relu, int beta) {
  __shared__ __bf16 As[2][BM * BK];
  __shared__ __bf16 Bs[2][BN * BK];

  const int tid  = threadIdx.x;
  const int lane = tid & 31;
  const int wave = tid >> 5;
  const int wm   = wave & 3;   // wave M offset = wm*32
  const int wn   = wave >> 2;  // wave N offset = wn*64
  const int m0   = blockIdx.x * BM;
  const int n0   = blockIdx.y * BN;

  // staging coords: 16B chunks; 512 chunks per 128x32 tile -> 2 per thread
  const int srow = tid >> 2;         // 0..63 (rows srow and srow+64)
  const int scol = (tid & 3) * 8;    // bf16 column offset (16B granule)

  auto stage = [&](int b, int k0) {
#if HAS_ASYNC_LDS
#pragma unroll
    for (int i = 0; i < 2; ++i) {
      int r  = srow + i * 64;
      int gr = m0 + r;
      if (gr >= M) gr = M - 1;  // clamp: uniform async count, rows masked later
      const __bf16* g = A + (size_t)gr * K + k0 + scol;
      __builtin_amdgcn_global_load_async_to_lds_b128(
          AS1V4(g), AS3V4(&As[b][r * BK + scol]), 0, 0);
    }
#pragma unroll
    for (int i = 0; i < 2; ++i) {
      int r = srow + i * 64;
      const __bf16* g = W + (size_t)(n0 + r) * K + k0 + scol;
      __builtin_amdgcn_global_load_async_to_lds_b128(
          AS1V4(g), AS3V4(&Bs[b][r * BK + scol]), 0, 0);
    }
#else
    uint4 va[2], vb[2];
#pragma unroll
    for (int i = 0; i < 2; ++i) {
      int gr = m0 + srow + i * 64;
      if (gr >= M) gr = M - 1;
      va[i] = *(const uint4*)(A + (size_t)gr * K + k0 + scol);
      vb[i] = *(const uint4*)(W + (size_t)(n0 + srow + i * 64) * K + k0 + scol);
    }
#pragma unroll
    for (int i = 0; i < 2; ++i) {
      *(uint4*)&As[b][(srow + i * 64) * BK + scol] = va[i];
      *(uint4*)&Bs[b][(srow + i * 64) * BK + scol] = vb[i];
    }
#endif
  };

  const int rowHalf = lane & 15;
  const int kb      = (lane >> 4) * 8;   // 16-bit fragment K-half per ISA layout

  v8f acc[2][4] = {};
  const int T = K / BK;

  stage(0, 0);
  for (int kt = 0; kt < T; ++kt) {
    int cur = kt & 1;
    if (kt + 1 < T) stage(cur ^ 1, (kt + 1) * BK);
#if HAS_ASYNC_LDS
    if (kt + 1 < T) __builtin_amdgcn_s_wait_asynccnt(4);   // tile kt done (in-order)
    else            __builtin_amdgcn_s_wait_asynccnt(0);
#endif
    __syncthreads();

    union FragU { v16bf v; uint4 q[2]; };
    FragU fa[2], fb[4];
#pragma unroll
    for (int mi = 0; mi < 2; ++mi) {
      int ar = wm * 32 + mi * 16 + rowHalf;
      fa[mi].q[0] = *(const uint4*)&As[cur][ar * BK + kb];
      fa[mi].q[1] = *(const uint4*)&As[cur][ar * BK + kb + 16];
    }
#pragma unroll
    for (int ni = 0; ni < 4; ++ni) {
      int br = wn * 64 + ni * 16 + rowHalf;
      fb[ni].q[0] = *(const uint4*)&Bs[cur][br * BK + kb];
      fb[ni].q[1] = *(const uint4*)&Bs[cur][br * BK + kb + 16];
    }
#pragma unroll
    for (int mi = 0; mi < 2; ++mi)
#pragma unroll
      for (int ni = 0; ni < 4; ++ni)
        acc[mi][ni] = __builtin_amdgcn_wmma_f32_16x16x32_bf16(
            false, fa[mi].v, false, fb[ni].v, (short)0, acc[mi][ni], false, false);

    __syncthreads();
  }

  // epilogue: C/D layout: VGPR r -> M=r (lanes 0-15) or M=r+8 (lanes 16-31), N=lane&15
#pragma unroll
  for (int mi = 0; mi < 2; ++mi) {
    int rbase = m0 + wm * 32 + mi * 16 + ((lane >> 4) ? 8 : 0);
#pragma unroll
    for (int ni = 0; ni < 4; ++ni) {
      int nIdx = n0 + wn * 64 + ni * 16 + (lane & 15);
      float bv = bias ? bias[nIdx] : 0.0f;
#pragma unroll
      for (int r = 0; r < 8; ++r) {
        int gr = rbase + r;
        if (gr < M) {
          size_t idx = (size_t)gr * N + nIdx;
          float v = acc[mi][ni][r] + bv;
          if (beta) v += Cf[idx];
          if (relu) v = fmaxf(v, 0.0f);
          if (Cb) Cb[idx] = (__bf16)v;
          else    Cf[idx] = v;
        }
      }
    }
  }
}

// ---------------------------------------------------------------- launcher
extern "C" void kernel_launch(void* const* d_in, const int* in_sizes, int n_in,
                              void* d_out, int out_size, void* d_ws, size_t ws_size,
                              hipStream_t stream) {
  (void)in_sizes; (void)n_in; (void)out_size; (void)ws_size;

  const float* x_news = (const float*)d_in[0];
  const float* x_kw   = (const float*)d_in[1];
  const float* x_st   = (const float*)d_in[2];
  const int* hk_s  = (const int*)d_in[3];
  const int* hk_d  = (const int*)d_in[4];
  const int* rhk_s = (const int*)d_in[5];
  const int* rhk_d = (const int*)d_in[6];
  const int* hs_s  = (const int*)d_in[7];
  const int* hs_d  = (const int*)d_in[8];
  const int* rhs_s = (const int*)d_in[9];
  const int* rhs_d = (const int*)d_in[10];
  auto F = [&](int i) { return (const float*)d_in[i]; };

  // workspace carve-out
  char* ws = (char*)d_ws;
  size_t off = 0;
  auto carveF = [&](size_t elems) {
    size_t o = off; off += (elems * 4 + 255) & ~(size_t)255; return (float*)(ws + o);
  };
  auto carveB = [&](size_t elems) {
    size_t o = off; off += (elems * 2 + 255) & ~(size_t)255; return (__bf16*)(ws + o);
  };

  __bf16* xb_news = carveB((size_t)N_NEWS * D_NEWS);
  __bf16* xb_kw   = carveB((size_t)N_KW * D_KW);
  __bf16* xb_st   = carveB((size_t)N_ST * D_ST);
  __bf16* hb_news = carveB((size_t)N_NEWS * HDIM);
  __bf16* hb_kw   = carveB((size_t)N_KW * HDIM);
  __bf16* hb_st   = carveB((size_t)N_ST * HDIM);
  float*  o_news  = carveF((size_t)N_NEWS * HDIM);
  float*  o_kw    = carveF((size_t)N_KW * HDIM);
  float*  o_st    = carveF((size_t)N_ST * HDIM);
  __bf16* ob_news = carveB((size_t)N_NEWS * HDIM);
  __bf16* ob_kw   = carveB((size_t)N_KW * HDIM);
  __bf16* ob_st   = carveB((size_t)N_ST * HDIM);
  float*  agg     = carveF((size_t)N_NEWS * HDIM);
  __bf16* aggb    = carveB((size_t)N_NEWS * HDIM);
  float*  cnt     = carveF((size_t)N_NEWS);

  float* out_news = (float*)d_out;
  float* out_kw   = out_news + (size_t)N_NEWS * ODIM;
  float* out_st   = out_kw   + (size_t)N_KW   * ODIM;

  auto cvt = [&](const float* in, __bf16* out, long n) {
    gnn_cvt_bf16<<<(int)((n + 255) / 256), 256, 0, stream>>>(in, out, n);
  };
  auto zero = [&](float* p, long n) {
    gnn_zero_f32<<<(int)((n + 255) / 256), 256, 0, stream>>>(p, n);
  };
  auto gemm = [&](const __bf16* A, const __bf16* W, const float* b,
                  float* Cf, __bf16* Cb, int M, int N, int K, int relu, int beta) {
    dim3 grid((M + BM - 1) / BM, N / BN);
    gnn_gemm<<<grid, 256, 0, stream>>>(A, W, b, Cf, Cb, M, N, K, relu, beta);
  };

  // ---- convert inputs + weights to bf16
  cvt(x_news, xb_news, (long)N_NEWS * D_NEWS);
  cvt(x_kw,   xb_kw,   (long)N_KW * D_KW);
  cvt(x_st,   xb_st,   (long)N_ST * D_ST);

  __bf16* wb_news = carveB((size_t)HDIM * D_NEWS);
  __bf16* wb_kw   = carveB((size_t)HDIM * D_KW);
  __bf16* wb_st   = carveB((size_t)HDIM * D_ST);
  cvt(F(11), wb_news, (long)HDIM * D_NEWS);
  cvt(F(13), wb_kw,   (long)HDIM * D_KW);
  cvt(F(15), wb_st,   (long)HDIM * D_ST);

  __bf16 *w1l[4], *w1r[4], *w2l[4], *w2r[4];
  const int i1[4][2] = {{17, 19}, {20, 22}, {23, 25}, {26, 28}};
  const int i2[4][2] = {{29, 31}, {32, 34}, {35, 37}, {38, 40}};
  for (int r = 0; r < 4; ++r) {
    w1l[r] = carveB((size_t)HDIM * HDIM);
    w1r[r] = carveB((size_t)HDIM * HDIM);
    cvt(F(i1[r][0]), w1l[r], (long)HDIM * HDIM);
    cvt(F(i1[r][1]), w1r[r], (long)HDIM * HDIM);
  }
  for (int r = 0; r < 4; ++r) {
    w2l[r] = carveB((size_t)ODIM * HDIM);
    w2r[r] = carveB((size_t)ODIM * HDIM);
    cvt(F(i2[r][0]), w2l[r], (long)ODIM * HDIM);
    cvt(F(i2[r][1]), w2r[r], (long)ODIM * HDIM);
  }

  // one SAGE relation: Cdst (+)= lin_l(mean_agg(srcFeat)) + lin_r(rootFeat)
  auto relation = [&](const __bf16* srcFeat, const __bf16* rootFeat,
                      const int* es, const int* ed, int E, int nDst,
                      const __bf16* Wl, const float* bl, const __bf16* Wr,
                      float* Cdst, int Nout, int betaFirst) {
    zero(agg, (long)nDst * HDIM);
    zero(cnt, nDst);
    gnn_scatter<<<(E + 7) / 8, 256, 0, stream>>>(srcFeat, es, ed, agg, cnt, E);
    long tot = (long)nDst * HDIM;
    gnn_mean_cvt<<<(int)((tot + 255) / 256), 256, 0, stream>>>(agg, cnt, aggb, tot);
    gemm(aggb, Wl, bl, Cdst, nullptr, nDst, Nout, HDIM, 0, betaFirst);
    gemm(rootFeat, Wr, nullptr, Cdst, nullptr, nDst, Nout, HDIM, 0, 1);
  };

  // ---- stage 1: type projections + ReLU, bf16 outputs
  gemm(xb_news, wb_news, F(12), nullptr, hb_news, N_NEWS, HDIM, D_NEWS, 1, 0);
  gemm(xb_kw,   wb_kw,   F(14), nullptr, hb_kw,   N_KW,   HDIM, D_KW,   1, 0);
  gemm(xb_st,   wb_st,   F(16), nullptr, hb_st,   N_ST,   HDIM, D_ST,   1, 0);

  // ---- conv1 (H -> H), f32 accumulation
  relation(hb_news, hb_kw,   hk_s,  hk_d,  E_HK, N_KW,   w1l[0], F(18), w1r[0], o_kw,   HDIM, 0);
  relation(hb_kw,   hb_news, rhk_s, rhk_d, E_HK, N_NEWS, w1l[1], F(21), w1r[1], o_news, HDIM, 0);
  relation(hb_news, hb_st,   hs_s,  hs_d,  E_HS, N_ST,   w1l[2], F(24), w1r[2], o_st,   HDIM, 0);
  relation(hb_st,   hb_news, rhs_s, rhs_d, E_HS, N_NEWS, w1l[3], F(27), w1r[3], o_news, HDIM, 1);

  {
    long n;
    n = (long)N_NEWS * HDIM; gnn_relu_cvt<<<(int)((n + 255) / 256), 256, 0, stream>>>(o_news, ob_news, n);
    n = (long)N_KW   * HDIM; gnn_relu_cvt<<<(int)((n + 255) / 256), 256, 0, stream>>>(o_kw,   ob_kw,   n);
    n = (long)N_ST   * HDIM; gnn_relu_cvt<<<(int)((n + 255) / 256), 256, 0, stream>>>(o_st,   ob_st,   n);
  }

  // ---- conv2 (H -> OUT) straight into d_out (f32)
  relation(ob_news, ob_kw,   hk_s,  hk_d,  E_HK, N_KW,   w2l[0], F(30), w2r[0], out_kw,   ODIM, 0);
  relation(ob_kw,   ob_news, rhk_s, rhk_d, E_HK, N_NEWS, w2l[1], F(33), w2r[1], out_news, ODIM, 0);
  relation(ob_news, ob_st,   hs_s,  hs_d,  E_HS, N_ST,   w2l[2], F(36), w2r[2], out_st,   ODIM, 0);
  relation(ob_st,   ob_news, rhs_s, rhs_d, E_HS, N_NEWS, w2l[3], F(39), w2r[3], out_news, ODIM, 1);
}
